// CCM_45483703665120
// MI455X (gfx1250) — compile-verified, matrix operand order checked by reading, not run on hardware
//
#include <hip/hip_runtime.h>

// ---------------------------------------------------------------------------
// Fused VMamba SS2D block for MI455X (gfx1250, wave32).
// One workgroup (256 threads = 8 waves) per batch image. All intermediates in
// LDS (~112KB of the 320KB/WGP). WMMA f16->f32 for the two projections, f32
// scalar scan with ds_add_f32 direction merge. Fragment traffic uses 16B
// ds_load_b128/ds_store_b128; tile selection is scalarized (readfirstlane).
// ---------------------------------------------------------------------------

typedef _Float16 f16x16 __attribute__((ext_vector_type(16)));
typedef _Float16 f16x8  __attribute__((ext_vector_type(8)));
typedef float    f32x8  __attribute__((ext_vector_type(8)));
typedef float    f32x4  __attribute__((ext_vector_type(4)));

#define LN_EPS 1e-5f

__device__ __forceinline__ float fsigmoid(float x){ return 1.f/(1.f+__expf(-x)); }
__device__ __forceinline__ float fsilu(float x){ return x*fsigmoid(x); }
__device__ __forceinline__ float fsoftplus(float x){ return (x>20.f)?x:__logf(1.f+__expf(x)); }

// scan-order index t -> hw-order position for direction k (self-inverse map;
// identical for gathering xs and scattering ys, see ref's wh_to_hw / ::-1).
__device__ __forceinline__ int dirmap(int k, int t){
  switch(k&3){
    case 0: return t;                               // hw
    case 1: return ((t&7)<<3)|(t>>3);               // wh transpose (H=W=8)
    case 2: return 63-t;                            // hw reversed
    default:{ int u=63-t; return ((u&7)<<3)|(u>>3); } // wh reversed
  }
}

struct Params {
  const float* in[36];
  float* out;
};

struct SMem {
  float x[64], xn[64], xD1[64], xD[64], x1[64];
  float wsum[9], bcs[9];                    // collapsed conv1x1->conv3x3 front
  alignas(16) float t[64][32];              // token embeddings (residual src)
  alignas(16) _Float16 u16[64][32];         // LN1 output, A-matrix of GEMM1
  alignas(16) _Float16 w_in16[128][32];     // B-matrix of GEMM1 (w_in[c][d])
  alignas(16) _Float16 w_out16[32][64];     // B-matrix of GEMM2 (w_out[o][i])
  alignas(16) float z[64][64];              // gate branch of in-projection
  alignas(16) float xmT[64][64];            // xm channel-major (DI, L)
  alignas(16) float xc[64][64];             // silu(dwconv(xm))   (DI, L)
  alignas(16) float dbc[4][6][64];          // x_dbl: [k][{dt0,dt1,B0,B1,C0,C1}][l]
  alignas(16) float y[64][64];              // direction-merged scan out (DI,L)
  float mu[64], rv[64];                     // out-norm stats per token
  alignas(16) _Float16 g16[64][64];         // gated, normed A-matrix of GEMM2
  alignas(16) float o[64][32];              // out-projection result
  float xu[64], xu2[64];
  float red[4];
};

__launch_bounds__(256)
__global__ void vmamba_fused(Params P)
{
  __shared__ SMem s;
  const int b    = blockIdx.x;
  const int tid  = threadIdx.x;
  const int lane = tid & 31;
  // wave id as SGPR: makes tile indices & branches wave-uniform scalar code
  const int wv   = __builtin_amdgcn_readfirstlane(tid >> 5);

  const float* x_i     = P.in[0];
  const float* bn_gamma= P.in[1];  const float* bn_beta = P.in[2];
  const float* bn_mean = P.in[3];  const float* bn_var  = P.in[4];
  const float* w_c1    = P.in[5];  const float* b_c1    = P.in[6];
  const float* w_c3    = P.in[7];  const float* b_c3    = P.in[8];
  const float* w_conv3 = P.in[9];  const float* b_conv3 = P.in[10];
  const float* w_dw    = P.in[11]; const float* b_dw    = P.in[12];
  const float* w_pw    = P.in[13]; const float* b_pw    = P.in[14];
  const float* w_pe    = P.in[15]; const float* b_pe    = P.in[16];
  const float* w_pu    = P.in[17]; const float* b_pu    = P.in[18];
  const float* ln1_g   = P.in[19]; const float* ln1_b   = P.in[20];
  const float* w_in    = P.in[21];
  const float* w_cdw   = P.in[22]; const float* b_cdw   = P.in[23];
  const float* x_proj_w= P.in[24];
  const float* dt_w    = P.in[25]; const float* dt_b    = P.in[26];
  const float* A_log   = P.in[27]; const float* Dskip   = P.in[28];
  const float* on_g    = P.in[29]; const float* on_b    = P.in[30];
  const float* w_out   = P.in[31];
  const float* w_si    = P.in[32]; const float* b_si    = P.in[33];
  const float* w_ci    = P.in[34]; const float* b_ci    = P.in[35];

  // WMMA fragment loader: 16-bit A/B 16x32 layout (ISA 7.12.2, wave32):
  // lane<16 holds K={0..7,16..23}, lane>=16 holds K={8..15,24..31}, row=lane&15.
  // Each half of the fragment is 16 contiguous bytes -> two ds_load_b128.
  auto loadFrag = [&](const _Float16* base, int stride, int row, int koff)->f16x16 {
    const int kb = (lane>>4)*8;
    const _Float16* p = base + row*stride + koff + kb;
    f16x8 lo = *(const f16x8*)(p);
    f16x8 hi = *(const f16x8*)(p + 16);
    f16x16 a;
#pragma unroll
    for (int j=0;j<8;++j){ a[j] = lo[j]; a[8+j] = hi[j]; }
    return a;
  };

  // ---- Phase 0: stage image + f16 weights, collapse conv1x1->conv3x3 ------
  if (tid < 64) s.x[tid] = x_i[b*64 + tid];
  for (int i=tid;i<128*32;i+=256) (&s.w_in16[0][0])[i]  = (_Float16)w_in[i];
  for (int i=tid;i<32*64;i+=256)  (&s.w_out16[0][0])[i] = (_Float16)w_out[i];
  if (tid < 9){
    float ws=0.f, bs=0.f;
    for (int c=0;c<32;++c){ float wc3=w_c3[c*9+tid]; ws += wc3*w_c1[c]; bs += wc3*b_c1[c]; }
    s.wsum[tid]=ws; s.bcs[tid]=bs;
  }
  __syncthreads();

  // ---- Phase 1: BN + dw/pw branch -----------------------------------------
  if (tid < 64){
    const int p=tid, h=p>>3, w=p&7;
    const float rbn = rsqrtf(bn_var[0]+LN_EPS);
    s.xn[p] = (s.x[p]-bn_mean[0])*rbn*bn_gamma[0] + bn_beta[0];
    float acc = b_dw[0];
#pragma unroll
    for (int dy=-1;dy<=1;++dy)
#pragma unroll
      for (int dx=-1;dx<=1;++dx){
        int hh=h+dy, ww=w+dx;
        if ((unsigned)hh<8u && (unsigned)ww<8u)
          acc += w_dw[(dy+1)*3+(dx+1)] * s.x[hh*8+ww];
      }
    s.xD1[p] = w_pw[0]*acc + b_pw[0];
  }
  __syncthreads();
  if (tid < 64){
    const int p=tid, h=p>>3, w=p&7;
    float asi = b_si[0], a1 = b_c3[0];
#pragma unroll
    for (int dy=-1;dy<=1;++dy)
#pragma unroll
      for (int dx=-1;dx<=1;++dx){
        int hh=h+dy, ww=w+dx;
        if ((unsigned)hh<8u && (unsigned)ww<8u){
          int q=hh*8+ww, tap=(dy+1)*3+(dx+1);
          asi += w_si[tap]*s.xD1[q];
          a1  += s.xn[q]*s.wsum[tap] + s.bcs[tap];
        }
      }
    s.xD[p] = fsigmoid(asi);
    s.x1[p] = a1;
  }
  __syncthreads();

  // ---- Phase 2: patch embed (outer product) -------------------------------
  for (int i=tid;i<64*32;i+=256){
    int l=i>>5, d=i&31;
    s.t[l][d] = s.x1[l]*w_pe[d] + b_pe[d];
  }
  __syncthreads();

  // ---- Phase 3: LN1 over D=32 -> f16 A-matrix -----------------------------
  if (tid < 64){
    const int l=tid;
    float mu=0.f;
#pragma unroll
    for (int d=0;d<32;++d) mu += s.t[l][d];
    mu *= (1.f/32.f);
    float var=0.f;
#pragma unroll
    for (int d=0;d<32;++d){ float df=s.t[l][d]-mu; var += df*df; }
    float rs = rsqrtf(var*(1.f/32.f)+LN_EPS);
#pragma unroll
    for (int d=0;d<32;++d)
      s.u16[l][d] = (_Float16)((s.t[l][d]-mu)*rs*ln1_g[d] + ln1_b[d]);
  }
  __syncthreads();

  // ---- Phase 4: in-projection GEMM (64x32)x(32x128), WMMA, 4 tiles/wave ---
#pragma unroll
  for (int i=0;i<4;++i){
    const int tile = wv*4 + i;           // scalar (wv is SGPR)
    const int mt = tile>>3, nt = tile&7; // scalar tile coords
    const int row  = mt*16 + (lane&15);
    const int colg = nt*16 + (lane&15);
    f16x16 a  = loadFrag(&s.u16[0][0],   32, row,  0);
    f16x16 bf = loadFrag(&s.w_in16[0][0],32, colg, 0);
    f32x8 c = {0.f,0.f,0.f,0.f,0.f,0.f,0.f,0.f};
    c = __builtin_amdgcn_wmma_f32_16x16x32_f16(false, a, false, bf, (short)0, c, false, false);
    const int rbase = mt*16 + ((lane>>4)<<3);
    if (nt < 4){
      // xm half: 8 C elements are contiguous rows -> two ds_store_b128
      f32x4 lo = {c[0],c[1],c[2],c[3]};
      f32x4 hi = {c[4],c[5],c[6],c[7]};
      *(f32x4*)&s.xmT[colg][rbase]   = lo;
      *(f32x4*)&s.xmT[colg][rbase+4] = hi;
    } else {
      // gate half: column-strided scatter
#pragma unroll
      for (int r=0;r<8;++r) s.z[rbase+r][colg-64] = c[r];
    }
  }
  __syncthreads();

  // ---- Phase 5: depthwise conv 3x3 + SiLU (per channel) -------------------
  {
    const int d = tid>>2;
    const float* wc = w_cdw + d*9;
    const float bc = b_cdw[d];
#pragma unroll
    for (int i=0;i<16;++i){
      int p = ((tid&3)<<4) + i, h=p>>3, w=p&7;
      float acc = bc;
#pragma unroll
      for (int dy=-1;dy<=1;++dy)
#pragma unroll
        for (int dx=-1;dx<=1;++dx){
          int hh=h+dy, ww=w+dx;
          if ((unsigned)hh<8u && (unsigned)ww<8u)
            acc += wc[(dy+1)*3+(dx+1)] * s.xmT[d][hh*8+ww];
        }
      s.xc[d][p] = fsilu(acc);
    }
  }
  // zero scan accumulator while waiting
  for (int i=tid;i<64*64;i+=256) (&s.y[0][0])[i] = 0.f;
  __syncthreads();

  // ---- Phase 6: x_dbl = x_proj_w . xs  (k,l per thread, 6 outputs) --------
  {
    const int k = tid>>6, l = tid&63;
    const int m = dirmap(k,l);
    const float* xp = x_proj_w + k*6*64;
    float a0=0,a1=0,a2=0,a3=0,a4=0,a5=0;
    for (int d=0;d<64;++d){
      float xv = s.xc[d][m];
      a0 += xp[0*64+d]*xv; a1 += xp[1*64+d]*xv; a2 += xp[2*64+d]*xv;
      a3 += xp[3*64+d]*xv; a4 += xp[4*64+d]*xv; a5 += xp[5*64+d]*xv;
    }
    s.dbc[k][0][l]=a0; s.dbc[k][1][l]=a1; s.dbc[k][2][l]=a2;
    s.dbc[k][3][l]=a3; s.dbc[k][4][l]=a4; s.dbc[k][5][l]=a5;
  }
  __syncthreads();

  // ---- Phase 7: selective scan, one (k,d) recurrence per thread -----------
  {
    const int k = tid>>6, d = tid&63;
    const int kd = k*64+d;
    const float dw0 = dt_w[kd*2+0], dw1 = dt_w[kd*2+1], dtb = dt_b[kd];
    const float A0 = -__expf(A_log[kd*2+0]);
    const float A1 = -__expf(A_log[kd*2+1]);
    const float Dsk = Dskip[kd];
    float h0=0.f, h1=0.f;
    for (int t=0;t<64;++t){
      const int m = dirmap(k,t);
      const float xv = s.xc[d][m];
      float dtv = fsoftplus(dw0*s.dbc[k][0][t] + dw1*s.dbc[k][1][t] + dtb);
      const float B0=s.dbc[k][2][t], B1=s.dbc[k][3][t];
      const float C0=s.dbc[k][4][t], C1=s.dbc[k][5][t];
      h0 = __expf(dtv*A0)*h0 + dtv*B0*xv;
      h1 = __expf(dtv*A1)*h1 + dtv*B1*xv;
      atomicAdd(&s.y[d][m], h0*C0 + h1*C1 + Dsk*xv);   // ds_add_f32 merge
    }
  }
  __syncthreads();

  // ---- Phase 8: out-norm stats over DI=64 per token -----------------------
  if (tid < 64){
    const int l=tid;
    float mu=0.f;
    for (int d=0;d<64;++d) mu += s.y[d][l];
    mu *= (1.f/64.f);
    float var=0.f;
    for (int d=0;d<64;++d){ float df=s.y[d][l]-mu; var += df*df; }
    s.mu[l]=mu; s.rv[l]=rsqrtf(var*(1.f/64.f)+LN_EPS);
  }
  __syncthreads();

  // ---- Phase 9: norm * silu(z) gate -> f16 A-matrix of GEMM2 --------------
  for (int i=tid;i<64*64;i+=256){
    int l=i>>6, d=i&63;
    float v  = (s.y[d][l]-s.mu[l])*s.rv[l]*on_g[d] + on_b[d];
    float zv = s.z[l][d];
    s.g16[l][d] = (_Float16)(v * fsilu(zv));
  }
  __syncthreads();

  // ---- Phase 10: out-projection GEMM (64x64)x(64x32), 1 tile/wave, K=2x32 -
  {
    const int mt = wv>>1, nt = wv&1;     // scalar tile coords
    const int row  = mt*16 + (lane&15);
    const int colg = nt*16 + (lane&15);
    f32x8 c = {0.f,0.f,0.f,0.f,0.f,0.f,0.f,0.f};
#pragma unroll
    for (int kc=0;kc<2;++kc){
      f16x16 a  = loadFrag(&s.g16[0][0],    64, row,  kc*32);
      f16x16 bf = loadFrag(&s.w_out16[0][0],64, colg, kc*32);
      c = __builtin_amdgcn_wmma_f32_16x16x32_f16(false, a, false, bf, (short)0, c, false, false);
    }
    const int rbase = mt*16 + ((lane>>4)<<3);
#pragma unroll
    for (int r=0;r<8;++r) s.o[rbase+r][colg] = c[r];
  }
  __syncthreads();

  // ---- Phase 11: residual + unembed + conv3x3 residual --------------------
  if (tid < 64){
    const int p=tid;
    float acc = b_pu[0];
#pragma unroll
    for (int d=0;d<32;++d) acc += (s.t[p][d] + s.o[p][d]) * w_pu[d];
    s.xu[p] = acc + s.x[p];
  }
  __syncthreads();
  if (tid < 64){
    const int p=tid, h=p>>3, w=p&7;
    float acc = b_conv3[0];
#pragma unroll
    for (int dy=-1;dy<=1;++dy)
#pragma unroll
      for (int dx=-1;dx<=1;++dx){
        int hh=h+dy, ww=w+dx;
        if ((unsigned)hh<8u && (unsigned)ww<8u)
          acc += w_conv3[(dy+1)*3+(dx+1)] * s.xu[hh*8+ww];
      }
    s.xu2[p] = acc + s.xu[p];
  }
  __syncthreads();

  // ---- Phase 12: channel gate + final combine -----------------------------
  if (tid == 0){
    float m=0.f;
    for (int p=0;p<64;++p) m += s.xu2[p];
    m *= (1.f/64.f);
    s.red[0] = fsigmoid(w_ci[0]*m + b_ci[0]);
  }
  __syncthreads();
  if (tid < 64){
    const int p=tid;
    P.out[b*64+p] = s.xD1[p]*s.red[0] + s.xD[p]*s.xu2[p];
  }
}

extern "C" void kernel_launch(void* const* d_in, const int* in_sizes, int n_in,
                              void* d_out, int out_size, void* d_ws, size_t ws_size,
                              hipStream_t stream) {
  (void)n_in; (void)out_size; (void)d_ws; (void)ws_size;
  Params P;
  for (int i=0;i<36;++i) P.in[i] = (const float*)d_in[i];
  P.out = (float*)d_out;
  const int B = in_sizes[0] / 64;   // (B,1,8,8)
  vmamba_fused<<<B, 256, 0, stream>>>(P);
}
